// QwenMoeWrapperBaseline_44418551775976
// MI455X (gfx1250) — matrix operand order repaired and checked by reading, not compiled, not run on hardware
//
#include <hip/hip_runtime.h>
#include <hip/hip_bf16.h>
#include <math.h>

// Problem constants (match reference)
#define Tn 2048   // B*S tokens
#define Dn 2048   // hidden
#define En 8      // experts
#define Fn 1408   // moe intermediate
#define TOPK 2

#define NFT_GU 88    // f-tiles for gate/up (Fn/16)
#define NKB_GU 64    // k-steps over D (Dn/32)
#define NFT_DN 128   // d-tiles for down (Dn/16)
#define NKB_DN 44    // k-steps over F (Fn/32)

typedef __attribute__((ext_vector_type(16))) __bf16 v16bf;
typedef __attribute__((ext_vector_type(8)))  float  v8f;

static __device__ __forceinline__ __bf16 f2bf(float f) {
    unsigned int u = __float_as_uint(f);
    u += 0x7FFFu + ((u >> 16) & 1u);        // RNE to bf16
    unsigned short h = (unsigned short)(u >> 16);
    return __builtin_bit_cast(__bf16, h);
}

// ---------------- K0: zero per-expert cursors ----------------
__global__ void moe_zero_cursor(int* __restrict__ cursor) {
    if (threadIdx.x < En) cursor[threadIdx.x] = 0;
}

// ---------------- Kp: pack an [E,K,N] fp32 weight into bf16 B-fragment layout ----------
// dst fragment fi = ((e*nft + ft)*nkb + kb)*32 + L, 16 contiguous bf16 per fragment:
//   element i  <->  row K = kb*32 + (L/16)*8 + (i<8 ? i : i+8),  col N = ft*16 + (L%16)
__global__ __launch_bounds__(256)
void moe_pack_b(const float* __restrict__ src, __bf16* __restrict__ dst,
                int Kdim, int Ndim, int nft, int nkb) {
    const int per_e = nft * nkb * 32;
    const int total = En * per_e;
    for (int fi = blockIdx.x * blockDim.x + threadIdx.x; fi < total;
         fi += gridDim.x * blockDim.x) {
        const int e  = fi / per_e;
        int r        = fi - e * per_e;
        const int ft = r / (nkb * 32);
        r           -= ft * (nkb * 32);
        const int kb = r >> 5;
        const int L  = r & 31;
        const int k0 = kb * 32 + ((L >> 4) << 3);
        const int n  = ft * 16 + (L & 15);
        const float* sp = src + ((size_t)e * Kdim + k0) * Ndim + n;
        v16bf v;
        #pragma unroll
        for (int i = 0; i < 16; ++i) v[i] = f2bf(sp[(size_t)(i + (i & 8)) * Ndim]);
        *(v16bf*)(dst + ((size_t)fi << 4)) = v;
    }
}

// ---------------- K1: router (one wave32 per token) ----------------
__global__ __launch_bounds__(128)
void moe_router(const float* __restrict__ x, const float* __restrict__ gw,
                int* __restrict__ sel, float* __restrict__ tokw) {
    const int wave = threadIdx.x >> 5;
    const int lane = threadIdx.x & 31;
    const int t = blockIdx.x * 4 + wave;
    if (t >= Tn) return;
    const float* xr = x + (size_t)t * Dn;

    float lg[En];
    #pragma unroll
    for (int e = 0; e < En; ++e) {
        const float* w = gw + (size_t)e * Dn;
        float p = 0.f;
        for (int d = lane; d < Dn; d += 32) p += xr[d] * w[d];
        #pragma unroll
        for (int off = 16; off >= 1; off >>= 1) p += __shfl_xor(p, off, 32);
        lg[e] = p;
    }
    if (lane == 0) {
        float m = lg[0];
        #pragma unroll
        for (int e = 1; e < En; ++e) m = fmaxf(m, lg[e]);
        float pe[En];
        #pragma unroll
        for (int e = 0; e < En; ++e) pe[e] = expf(lg[e] - m); // denom cancels in renorm
        int i1 = 0; float b1 = pe[0];
        #pragma unroll
        for (int e = 1; e < En; ++e) if (pe[e] > b1) { b1 = pe[e]; i1 = e; }
        int i2 = (i1 == 0) ? 1 : 0; float b2 = pe[i2];
        #pragma unroll
        for (int e = 0; e < En; ++e)
            if (e != i1 && pe[e] > b2) { b2 = pe[e]; i2 = e; }
        const float den = b1 + b2;
        sel[t * 2 + 0] = i1;  tokw[t * 2 + 0] = b1 / den;
        sel[t * 2 + 1] = i2;  tokw[t * 2 + 1] = b2 / den;
    }
}

// ---------------- K2: scatter (token,k) pairs into expert buckets ----------------
__global__ void moe_scatter(const int* __restrict__ sel, int* __restrict__ cursor,
                            int* __restrict__ bucket) {
    const int i = blockIdx.x * blockDim.x + threadIdx.x;
    if (i >= Tn * TOPK) return;
    const int e = sel[i];
    const int pos = atomicAdd(&cursor[e], 1);
    bucket[e * Tn + pos] = i;    // packed (t*2 + k)
}

// ---------------- K3: grouped expert GEMM (WMMA bf16) ----------------
// A-fragment LDS layout: frag (kstep, L) at 16 contiguous bf16;
//   element i <-> K = (i<8 ? i : i+8) + (L>=16 ? 8 : 0), row M = L%16
__global__ __launch_bounds__(256)
void moe_expert(const float* __restrict__ x,
                const __bf16* __restrict__ wgB,
                const __bf16* __restrict__ wuB,
                const __bf16* __restrict__ wdB,
                const int*   __restrict__ cursor,
                const int*   __restrict__ bucket,
                const float* __restrict__ tokw,
                float*       __restrict__ pair) {
    __shared__ __bf16 XA[16 * 32 * 16];       // 16 k-steps of A fragments (16 KB)
    __shared__ __bf16 HA[NKB_DN * 32 * 16];   // 44 k-steps of H fragments (44 KB)
    __shared__ int    s_pair[16];
    __shared__ int    s_tok[16];
    __shared__ float  s_w[16];

    const int e    = blockIdx.y;
    const int tile = blockIdx.x;
    const int n    = cursor[e];
    if (tile * 16 >= n) return;               // uniform early exit
    const int rows = min(16, n - tile * 16);

    const int tid  = threadIdx.x;
    const int wave = tid >> 5;
    const int lane = tid & 31;
    const int lrow = lane & 15;
    const int lhi  = lane >> 4;

    if (tid < 16) {
        const int idx = tile * 16 + tid;
        const int ent = bucket[e * Tn + ((idx < n) ? idx : tile * 16)];
        s_pair[tid] = ent;
        s_tok[tid]  = ent >> 1;
        s_w[tid]    = (idx < n) ? tokw[ent] : 0.f;
    }
    __syncthreads();

    // ---- Phase 1: H = silu(X Wg) * (X Wu); 88 f-tiles = 11 rounds x 8 waves ----
    #pragma unroll 1
    for (int rnd = 0; rnd < 11; ++rnd) {
        const int ft = rnd * 8 + wave;
        const __bf16* wgF = wgB + ((size_t)(e * NFT_GU + ft) << 15); // 64*512 per tile
        const __bf16* wuF = wuB + ((size_t)(e * NFT_GU + ft) << 15);
        __builtin_prefetch(wgF, 0, 1);
        __builtin_prefetch(wuF, 0, 1);
        v8f accg = {}; v8f accu = {};
        #pragma unroll 1
        for (int kc = 0; kc < 4; ++kc) {                 // 4 chunks of 512 along D
            __syncthreads();
            // stage 16 A-fragment k-steps: thread -> (ks,L) pair, 32B LDS store
            #pragma unroll 1
            for (int p = tid; p < 512; p += 256) {
                const int ks = p >> 5, L = p & 31;
                const int koff = (L >> 4) << 3;          // 0 or 8
                const float* sp = x + (size_t)s_tok[L & 15] * Dn + kc * 512 + ks * 32 + koff;
                v16bf frag;
                #pragma unroll
                for (int i = 0; i < 8; ++i) {
                    frag[i]     = f2bf(sp[i]);
                    frag[8 + i] = f2bf(sp[16 + i]);
                }
                *(v16bf*)(XA + (p << 4)) = frag;
            }
            __syncthreads();
            #pragma unroll 4
            for (int ksl = 0; ksl < 16; ++ksl) {
                const int kb = kc * 16 + ksl;
                const v16bf a  = *(const v16bf*)(XA + ((ksl * 32 + lane) << 4));
                const v16bf bg = *(const v16bf*)(wgF + (((size_t)kb * 32 + lane) << 4));
                const v16bf bu = *(const v16bf*)(wuF + (((size_t)kb * 32 + lane) << 4));
                accg = __builtin_amdgcn_wmma_f32_16x16x32_bf16(
                    false, a, false, bg, (short)0, accg, false, false);
                accu = __builtin_amdgcn_wmma_f32_16x16x32_bf16(
                    false, a, false, bu, (short)0, accu, false, false);
            }
        }
        // fused SiLU(g)*u -> H stored directly in A-fragment layout for phase 2
        const int f   = ft * 16 + lrow;                   // this lane's H column
        const int fks = f >> 5;
        const int k32 = f & 31;
        const int hi  = (k32 >> 3) & 1;
        const int i0  = (k32 & 7) + ((k32 & 16) ? 8 : 0);
        #pragma unroll
        for (int r = 0; r < 8; ++r) {
            const int M = r + lhi * 8;
            const float g = accg[r];
            const float s = g * __builtin_amdgcn_rcpf(1.f + __expf(-g));
            HA[((fks * 32 + (M + (hi << 4))) << 4) + i0] = f2bf(s * accu[r]);
        }
    }
    __syncthreads();

    // ---- Phase 2: Out = (H * w_token) Wd; 128 d-tiles = 16 per wave ----
    #pragma unroll 1
    for (int jj = 0; jj < 16; ++jj) {
        const int dt = jj * 8 + wave;
        const __bf16* wdF = wdB + (size_t)(e * NFT_DN + dt) * (NKB_DN * 512);
        __builtin_prefetch(wdF, 0, 1);
        v8f acc = {};
        #pragma unroll 4
        for (int fk = 0; fk < NKB_DN; ++fk) {
            const v16bf a  = *(const v16bf*)(HA + ((fk * 32 + lane) << 4));
            const v16bf bd = *(const v16bf*)(wdF + (((size_t)fk * 32 + lane) << 4));
            acc = __builtin_amdgcn_wmma_f32_16x16x32_bf16(
                false, a, false, bd, (short)0, acc, false, false);
        }
        const int d0 = dt * 16;
        #pragma unroll
        for (int r = 0; r < 8; ++r) {
            const int M = r + lhi * 8;
            if (M < rows)
                pair[(size_t)s_pair[M] * Dn + d0 + lrow] = acc[r] * s_w[M];
        }
    }
}

// ---------------- K4: combine the two expert contributions per token ----------------
__global__ void moe_combine(const float* __restrict__ pair, float* __restrict__ out) {
    const size_t total = (size_t)Tn * Dn;
    for (size_t i = (size_t)blockIdx.x * blockDim.x + threadIdx.x; i < total;
         i += (size_t)gridDim.x * blockDim.x) {
        const size_t t = i / Dn, d = i % Dn;
        out[i] = pair[(t * 2 + 0) * Dn + d] + pair[(t * 2 + 1) * Dn + d];
    }
}

// ---------------- launcher ----------------
extern "C" void kernel_launch(void* const* d_in, const int* in_sizes, int n_in,
                              void* d_out, int out_size, void* d_ws, size_t ws_size,
                              hipStream_t stream) {
    const float* x      = (const float*)d_in[0];   // [B,S,D]
    const float* gate_w = (const float*)d_in[1];   // [E,D]
    const float* w_gate = (const float*)d_in[2];   // [E,D,F]
    const float* w_up   = (const float*)d_in[3];   // [E,D,F]
    const float* w_down = (const float*)d_in[4];   // [E,F,D]
    float* out = (float*)d_out;

    // workspace layout
    size_t o = 0;
    char* ws = (char*)d_ws;
    int*   cursor = (int*)(ws + o);   o += 8 * sizeof(int);
    int*   sel    = (int*)(ws + o);   o += (size_t)Tn * 2 * sizeof(int);
    int*   bucket = (int*)(ws + o);   o += (size_t)En * Tn * sizeof(int);
    float* tokw   = (float*)(ws + o); o += (size_t)Tn * 2 * sizeof(float);
    float* pair   = (float*)(ws + o); o += (size_t)Tn * 2 * Dn * sizeof(float);
    o = (o + 31) & ~(size_t)31;
    const size_t NPK = (size_t)En * Dn * Fn;       // elements per packed matrix
    __bf16* wgB = (__bf16*)(ws + o);  o += NPK * 2;
    __bf16* wuB = (__bf16*)(ws + o);  o += NPK * 2;
    __bf16* wdB = (__bf16*)(ws + o);  o += NPK * 2;

    moe_zero_cursor<<<1, 32, 0, stream>>>(cursor);
    moe_pack_b<<<5632, 256, 0, stream>>>(w_gate, wgB, Dn, Fn, NFT_GU, NKB_GU);
    moe_pack_b<<<5632, 256, 0, stream>>>(w_up,   wuB, Dn, Fn, NFT_GU, NKB_GU);
    moe_pack_b<<<5632, 256, 0, stream>>>(w_down, wdB, Fn, Dn, NFT_DN, NKB_DN);
    moe_router<<<Tn / 4, 128, 0, stream>>>(x, gate_w, sel, tokw);
    moe_scatter<<<(Tn * TOPK + 255) / 256, 256, 0, stream>>>(sel, cursor, bucket);
    moe_expert<<<dim3(Tn / 16, En), 256, 0, stream>>>(
        x, wgB, wuB, wdB, cursor, bucket, tokw, pair);
    moe_combine<<<1024, 256, 0, stream>>>(pair, out);
}
